// SemiAR_Decoder_65068754534593
// MI455X (gfx1250) — matrix operand order repaired and checked by reading, not compiled, not run on hardware
//
#include <hip/hip_runtime.h>

// ---------------- problem constants ----------------
#define N_SEQ   16384
#define HID     256
#define DD      64
#define LYV     48
#define NOUTV   4
#define NIT     12
#define MTILE   32
#define NBLK    (N_SEQ / MTILE)      // 512 blocks per scan

// packed weight geometry (K padded to multiples of 32, N to multiples of 16)
#define KC_IH   5      // GRU Wih: K 129 -> 160
#define KC_HH   8      // GRU Whh: K 256
#define KC_W1   21     // MLP W1: K 641 -> 672 (13 chunks xm-seg + 8 chunks h)
#define KC_W2   8
#define KC_W3   8
#define NT_G    48     // 768/16
#define NT_H    16     // 256/16

// LDS row strides (bf16 elements; +8 pad to dodge bank conflicts, keeps 16B align)
#define S_H     264    // hidden state 256 + 8
#define S_XG    168    // GRU input 160 + 8
#define S_XM    424    // MLP input seg1 416 + 8
#define S_A     264    // activations 256 + 8

typedef __bf16 bf16_t;
typedef __bf16 v16bf __attribute__((ext_vector_type(16)));
typedef __bf16 v8bf  __attribute__((ext_vector_type(8)));
typedef float  v8f   __attribute__((ext_vector_type(8)));

__device__ __forceinline__ unsigned short f2bf_bits(float f) {
  unsigned u = __builtin_bit_cast(unsigned, f);
  unsigned r = u + 0x7FFFu + ((u >> 16) & 1u);   // round-to-nearest-even
  return (unsigned short)(r >> 16);
}
__device__ __forceinline__ bf16_t f2bf(float f) {
  unsigned short s = f2bf_bits(f);
  return __builtin_bit_cast(bf16_t, s);
}
__device__ __forceinline__ float bf2f(bf16_t b) {
  unsigned short s = __builtin_bit_cast(unsigned short, b);
  return __builtin_bit_cast(float, ((unsigned)s) << 16);
}
__device__ __forceinline__ float sigm(float x) { return 1.f / (1.f + __expf(-x)); }

// One 16x16x32 bf16 WMMA: A from row-major LDS (lane ptr pre-offset by
// (m0+ln)*stride + hi*8), B from packed global fragment (lane*16 elems).
__device__ __forceinline__ v8f wmma_bf(const bf16_t* aPtr, const bf16_t* bPtr, v8f acc) {
  v8bf a0 = *(const v8bf*)(aPtr);        // lanes0-15: K 0..7  | lanes16-31: K 8..15
  v8bf a1 = *(const v8bf*)(aPtr + 16);   // lanes0-15: K16..23 | lanes16-31: K24..31
  v16bf a = __builtin_shufflevector(a0, a1, 0,1,2,3,4,5,6,7,8,9,10,11,12,13,14,15);
  v16bf b = *(const v16bf*)(bPtr);       // 32B contiguous per lane
  return __builtin_amdgcn_wmma_f32_16x16x32_bf16(false, a, false, b, (short)0, acc, false, false);
}

// ---------------- prep: pack W^T into B-fragment order (bf16) ----------------
// dst layout: frag(nt,kc) at ((nt*KC+kc)*512) bf16; within frag: lane*16 + v*2 + h
// element (lane,v,h): k = kc*32 + (lane>>4)*8 + 2v + h + (v>=4 ? 8 : 0) ; n = nt*16 + (lane&15)
// mode 0: src k if k<Ksrc else 0.  mode 2 (W1 remap): k<129 -> k ; 129..159 -> 0 ; k>=160 -> k-31
extern "C" __global__ void __launch_bounds__(256)
pack_weight(bf16_t* __restrict__ dst, const float* __restrict__ src,
            int NT, int KC, int Nsrc, int Ksrc, int srcStride, int mode)
{
  int total = NT * KC * 256;             // dwords
  int i = blockIdx.x * 256 + threadIdx.x;
  if (i >= total) return;
  int frag = i >> 8, within = i & 255;
  int lane = within >> 3, v = within & 7;
  int kc = frag % KC, nt = frag / KC;
  int n = nt * 16 + (lane & 15);
  int kbase = kc * 32 + (lane >> 4) * 8 + v * 2 + (v >= 4 ? 8 : 0);
  unsigned outw = 0;
  for (int h = 0; h < 2; ++h) {
    int k = kbase + h;
    int ks;
    if (mode == 2) ks = (k < 129) ? k : ((k >= 160) ? (k - 31) : -1);
    else           ks = (k < Ksrc) ? k : -1;
    float val = 0.f;
    if (ks >= 0 && ks < Ksrc && n < Nsrc) val = src[(size_t)n * srcStride + ks];
    outw |= ((unsigned)f2bf_bits(val)) << (16 * h);
  }
  ((unsigned*)dst)[i] = outw;
}

// ---------------- prep: ep and dt_truth from Y ----------------
extern "C" __global__ void __launch_bounds__(256)
prep_y(const float* __restrict__ Y, float* __restrict__ ep, float* __restrict__ dtt)
{
  int i = blockIdx.x * 256 + threadIdx.x;
  if (i >= N_SEQ * NIT) return;
  int n = i / NIT, it = i % NIT;
  const float* yb = Y + ((size_t)n * LYV + it * NOUTV) * 3;
  ep[i]  = yb[0] + yb[3] + yb[6] + yb[9];   // sum of Y[...,0] over NOUT group
  dtt[i] = yb[2];                            // Y[:, ::NOUT, 2]
}

// ---------------- main persistent recurrent kernel ----------------
extern "C" __global__ void __launch_bounds__(256, 4)
semiar_main(const float* __restrict__ TE_w, const float* __restrict__ shared_w,
            const float* __restrict__ shared_b, const float* __restrict__ TE_lam,
            const float* __restrict__ bih, const float* __restrict__ bhh,
            const float* __restrict__ pb1, const float* __restrict__ pb2, const float* __restrict__ pb3,
            const float* __restrict__ fb1, const float* __restrict__ fb2, const float* __restrict__ fb3,
            const float* __restrict__ h_int, const float* __restrict__ dt_init,
            const bf16_t* __restrict__ Wih_pk, const bf16_t* __restrict__ Whh_pk,
            const bf16_t* __restrict__ pW1_pk, const bf16_t* __restrict__ pW2_pk, const bf16_t* __restrict__ pW3_pk,
            const bf16_t* __restrict__ fW1_pk, const bf16_t* __restrict__ fW2_pk, const bf16_t* __restrict__ fW3_pk,
            const float* __restrict__ ep, const float* __restrict__ dtt,
            float* __restrict__ out)
{
  extern __shared__ char smem[];
  bf16_t* hA  = (bf16_t*)smem;              // [2][32][S_H] double-buffered hidden (bf16)
  bf16_t* xg  = hA  + 2 * MTILE * S_H;      // [32][S_XG] GRU input  (elapsed | te)
  bf16_t* xm  = xg  + MTILE * S_XG;         // [32][S_XM] MLP seg1   (dt | te | pad | h_int)
  bf16_t* a1s = xm  + MTILE * S_XM;         // [32][S_A]  MLP1 activations
  bf16_t* a2s = a1s + MTILE * S_A;          // [32][S_A]  MLP2 activations
  float*  teW = (float*)(a2s + MTILE * S_A);// [32][DD]
  float*  shw = teW + MTILE * DD;           // [DD]
  float*  shb = shw + DD;                   // [DD]
  float*  scDt = shb + DD;                  // [32] dt_pred / dt_truth
  float*  scEl = scDt + MTILE;              // [32] elapsed / et
  float*  pf  = scEl + MTILE;               // [32][16] MLP3 raw output

  const bool isFlow = (blockIdx.x >= NBLK);
  const int  mb     = blockIdx.x & (NBLK - 1);
  const int  row0   = mb * MTILE;
  const int  t      = threadIdx.x;
  const int  lane   = t & 31, wave = t >> 5;
  const int  ln = lane & 15, hi = lane >> 4;
  const int  mi = wave & 1, ngp = wave >> 1;      // 2 M-tiles x 4 N-groups
  const int  mrow0 = mi * 16;
  const int  bofs  = lane << 4;                   // per-lane B-fragment byte/elem offset

  const bf16_t* W1p = isFlow ? fW1_pk : pW1_pk;
  const bf16_t* W2p = isFlow ? fW2_pk : pW2_pk;
  const bf16_t* W3p = isFlow ? fW3_pk : pW3_pk;
  const float*  B1  = isFlow ? fb1 : pb1;
  const float*  B2  = isFlow ? fb2 : pb2;
  const float*  B3  = isFlow ? fb3 : pb3;

  // ---- one-time init ----
  for (int i = t; i < MTILE * DD; i += 256)
    teW[i] = TE_w[(size_t)(row0 + i / DD) * DD + (i % DD)];
  for (int i = t; i < DD; i += 256) { shw[i] = shared_w[i]; shb[i] = shared_b[i]; }
  for (int i = t; i < MTILE * HID; i += 256) {
    int r = i / HID, c = i % HID;
    bf16_t hv = f2bf(h_int[(size_t)(row0 + r) * HID + c]);
    hA[r * S_H + c]        = hv;           // ht(0) = h_t_int
    xm[r * S_XM + 160 + c] = hv;           // static h_t_int segment of MLP input
  }
  for (int i = t; i < MTILE * 31; i += 256) {          // zero K-pad cols 129..159
    int r = i / 31, c = 129 + i % 31;
    xg[r * S_XG + c] = f2bf(0.f);
    xm[r * S_XM + c] = f2bf(0.f);
  }
  if (t < MTILE) { float d0 = dt_init[row0 + t]; scDt[t] = d0; scEl[t] = d0; }
  const float lam = __expf(-TE_lam[0] * TE_lam[0]);
  __syncthreads();

  // per-lane A-fragment base offsets (row-major LDS)
  const int aoff = (mrow0 + ln);
  const bf16_t* aG  = xg  + aoff * S_XG + hi * 8;
  const bf16_t* aM  = xm  + aoff * S_XM + hi * 8;
  const bf16_t* aA1 = a1s + aoff * S_A  + hi * 8;
  const bf16_t* aA2 = a2s + aoff * S_A  + hi * 8;

  #pragma unroll 1
  for (int it = 0; it < NIT; ++it) {
    const int cur = it & 1, nxt = cur ^ 1;
    bf16_t* hAc = hA + cur * MTILE * S_H;
    bf16_t* hAn = hA + nxt * MTILE * S_H;
    const bf16_t* aH  = hAc + aoff * S_H + hi * 8;
    const bf16_t* aHn = hAn + aoff * S_H + hi * 8;

    // ---- stage 0: per-step scalars ----
    if (isFlow && t < MTILE) scDt[t] = dtt[(size_t)(row0 + t) * NIT + it];
    __syncthreads();

    // ---- stage 1: time encodings -> xg (from elapsed) and xm (from dt) ----
    for (int i = t; i < MTILE * DD; i += 256) {
      int r = i >> 6, d = i & 63;
      float el = scEl[r], dv = scDt[r];
      float w = teW[r * DD + d], sw = shw[d], sb = shb[d];
      float i1 = el * w, s1 = el * sw + sb;
      xg[r * S_XG + 1 + d]      = f2bf((1.f - lam) * __sinf(i1) + lam * __sinf(s1));
      xg[r * S_XG + 1 + DD + d] = f2bf((1.f - lam) * __cosf(i1) + lam * __cosf(s1));
      float i2 = dv * w, s2 = dv * sw + sb;
      xm[r * S_XM + 1 + d]      = f2bf((1.f - lam) * __sinf(i2) + lam * __sinf(s2));
      xm[r * S_XM + 1 + DD + d] = f2bf((1.f - lam) * __cosf(i2) + lam * __cosf(s2));
    }
    if (t < MTILE) { xg[t * S_XG] = f2bf(scEl[t]); xm[t * S_XM] = f2bf(scDt[t]); }
    __syncthreads();

    // ---- stage 2: GRU. Per hidden tile: 4 fused accumulator chains
    //      (r and z gates share one chain across Wih+Whh; n gate split) ----
    #pragma unroll 1
    for (int tt = 0; tt < 4; ++tt) {
      const int ct = ngp * 4 + tt, c0 = ct * 16;
      v8f ar  = {0,0,0,0,0,0,0,0};   // gi_r + gh_r
      v8f az  = {0,0,0,0,0,0,0,0};   // gi_z + gh_z
      v8f gin = {0,0,0,0,0,0,0,0};   // gi_n
      v8f ghn = {0,0,0,0,0,0,0,0};   // gh_n
      #pragma unroll
      for (int kc = 0; kc < KC_IH; ++kc) {
        const bf16_t* ap = aG + kc * 32;
        ar  = wmma_bf(ap, Wih_pk + ((size_t)(( 0 + ct) * KC_IH + kc) << 9) + bofs, ar);
        az  = wmma_bf(ap, Wih_pk + ((size_t)((16 + ct) * KC_IH + kc) << 9) + bofs, az);
        gin = wmma_bf(ap, Wih_pk + ((size_t)((32 + ct) * KC_IH + kc) << 9) + bofs, gin);
      }
      #pragma unroll
      for (int kc = 0; kc < KC_HH; ++kc) {
        const bf16_t* ap = aH + kc * 32;
        ar  = wmma_bf(ap, Whh_pk + ((size_t)(( 0 + ct) * KC_HH + kc) << 9) + bofs, ar);
        az  = wmma_bf(ap, Whh_pk + ((size_t)((16 + ct) * KC_HH + kc) << 9) + bofs, az);
        ghn = wmma_bf(ap, Whh_pk + ((size_t)((32 + ct) * KC_HH + kc) << 9) + bofs, ghn);
      }
      const float br = bih[c0 + ln] + bhh[c0 + ln];
      const float bz = bih[HID + c0 + ln] + bhh[HID + c0 + ln];
      const float bn = bih[2 * HID + c0 + ln];
      const float cn = bhh[2 * HID + c0 + ln];
      #pragma unroll
      for (int v = 0; v < 8; ++v) {
        int m = mrow0 + v + 8 * hi;
        float rg = sigm(ar[v] + br);
        float zg = sigm(az[v] + bz);
        float nn = tanhf(gin[v] + bn + rg * (ghn[v] + cn));
        float hp = bf2f(hAc[m * S_H + c0 + ln]);
        hAn[m * S_H + c0 + ln] = f2bf((1.f - zg) * nn + zg * hp);
      }
    }
    __syncthreads();

    // ---- stage 3: MLP layer 1 (K segments: xm 13 chunks + new h 8 chunks) ----
    #pragma unroll 1
    for (int tt = 0; tt < 4; ++tt) {
      const int nt = ngp * 4 + tt;
      v8f acc = {0,0,0,0,0,0,0,0};
      #pragma unroll
      for (int kc = 0; kc < 13; ++kc)
        acc = wmma_bf(aM + kc * 32, W1p + ((size_t)(nt * KC_W1 + kc) << 9) + bofs, acc);
      #pragma unroll
      for (int kc = 0; kc < 8; ++kc)
        acc = wmma_bf(aHn + kc * 32, W1p + ((size_t)(nt * KC_W1 + 13 + kc) << 9) + bofs, acc);
      const float b = B1[nt * 16 + ln];
      #pragma unroll
      for (int v = 0; v < 8; ++v) {
        float x = acc[v] + b; x = x > 0.f ? x : 0.f;
        a1s[(mrow0 + v + 8 * hi) * S_A + nt * 16 + ln] = f2bf(x);
      }
    }
    __syncthreads();

    // ---- stage 4: MLP layer 2 ----
    #pragma unroll 1
    for (int tt = 0; tt < 4; ++tt) {
      const int nt = ngp * 4 + tt;
      v8f acc = {0,0,0,0,0,0,0,0};
      #pragma unroll
      for (int kc = 0; kc < KC_W2; ++kc)
        acc = wmma_bf(aA1 + kc * 32, W2p + ((size_t)(nt * KC_W2 + kc) << 9) + bofs, acc);
      const float b = B2[nt * 16 + ln];
      #pragma unroll
      for (int v = 0; v < 8; ++v) {
        float x = acc[v] + b; x = x > 0.f ? x : 0.f;
        a2s[(mrow0 + v + 8 * hi) * S_A + nt * 16 + ln] = f2bf(x);
      }
    }
    __syncthreads();

    // ---- stage 5: MLP layer 3 (N padded 4->16; only one n-tile) ----
    if (ngp == 0) {
      v8f acc = {0,0,0,0,0,0,0,0};
      #pragma unroll
      for (int kc = 0; kc < KC_W3; ++kc)
        acc = wmma_bf(aA2 + kc * 32, W3p + ((size_t)kc << 9) + bofs, acc);
      const float b = (ln < NOUTV) ? B3[ln] : 0.f;
      #pragma unroll
      for (int v = 0; v < 8; ++v)
        pf[(mrow0 + v + 8 * hi) * 16 + ln] = acc[v] + b;
    }
    __syncthreads();

    // ---- stage 6: outputs + carry update ----
    if (t < MTILE) {
      const int grow = row0 + t;
      if (!isFlow) {
        float s = 0.f;
        #pragma unroll
        for (int j = 0; j < NOUTV; ++j) {
          float p = sigm(pf[t * 16 + j]);
          out[(size_t)grow * LYV + it * NOUTV + j] = p;
          s += p;
        }
        scDt[t] += s;       // dt_pred += sum(periods)
        scEl[t]  = s;       // elapsed = sum(periods)
      } else {
        #pragma unroll
        for (int j = 0; j < NOUTV; ++j)
          out[(size_t)N_SEQ * LYV + (size_t)grow * LYV + it * NOUTV + j] = pf[t * 16 + j];
        scEl[t] = ep[(size_t)grow * NIT + it];   // et_next = ep(this step)
      }
    }
    __syncthreads();
  }
}

// ---------------- host launcher ----------------
extern "C" void kernel_launch(void* const* d_in, const int* in_sizes, int n_in,
                              void* d_out, int out_size, void* d_ws, size_t ws_size,
                              hipStream_t stream) {
  const float* TE_w     = (const float*)d_in[0];
  const float* shared_w = (const float*)d_in[1];
  const float* shared_b = (const float*)d_in[2];
  const float* TE_lam   = (const float*)d_in[3];
  const float* gru_Wih  = (const float*)d_in[4];
  const float* gru_Whh  = (const float*)d_in[5];
  const float* gru_bih  = (const float*)d_in[6];
  const float* gru_bhh  = (const float*)d_in[7];
  const float* pW1      = (const float*)d_in[8];
  const float* pb1      = (const float*)d_in[9];
  const float* pW2      = (const float*)d_in[10];
  const float* pb2      = (const float*)d_in[11];
  const float* pW3      = (const float*)d_in[12];
  const float* pb3      = (const float*)d_in[13];
  const float* fW1      = (const float*)d_in[14];
  const float* fb1      = (const float*)d_in[15];
  const float* fW2      = (const float*)d_in[16];
  const float* fb2      = (const float*)d_in[17];
  const float* fW3      = (const float*)d_in[18];
  const float* fb3      = (const float*)d_in[19];
  const float* h_int    = (const float*)d_in[20];
  const float* dt_init  = (const float*)d_in[21];
  const float* Y        = (const float*)d_in[23];
  float* out            = (float*)d_out;
  (void)in_sizes; (void)n_in; (void)out_size; (void)ws_size;

  // workspace carve-out (bytes)
  char* ws = (char*)d_ws;
  size_t o = 0;
  bf16_t* Wih_pk = (bf16_t*)(ws + o); o += (size_t)NT_G * KC_IH * 512 * 2;   // 245760
  bf16_t* Whh_pk = (bf16_t*)(ws + o); o += (size_t)NT_G * KC_HH * 512 * 2;   // 393216
  bf16_t* pW1_pk = (bf16_t*)(ws + o); o += (size_t)NT_H * KC_W1 * 512 * 2;   // 344064
  bf16_t* pW2_pk = (bf16_t*)(ws + o); o += (size_t)NT_H * KC_W2 * 512 * 2;   // 131072
  bf16_t* pW3_pk = (bf16_t*)(ws + o); o += (size_t)1    * KC_W3 * 512 * 2;   // 8192
  bf16_t* fW1_pk = (bf16_t*)(ws + o); o += (size_t)NT_H * KC_W1 * 512 * 2;
  bf16_t* fW2_pk = (bf16_t*)(ws + o); o += (size_t)NT_H * KC_W2 * 512 * 2;
  bf16_t* fW3_pk = (bf16_t*)(ws + o); o += (size_t)1    * KC_W3 * 512 * 2;
  float*  epw    = (float*)(ws + o);  o += (size_t)N_SEQ * NIT * 4;
  float*  dttw   = (float*)(ws + o);  o += (size_t)N_SEQ * NIT * 4;

  auto packL = [&](bf16_t* dst, const float* src, int NT, int KC, int Nsrc, int Ksrc, int st, int mode) {
    int total = NT * KC * 256;
    pack_weight<<<(total + 255) / 256, 256, 0, stream>>>(dst, src, NT, KC, Nsrc, Ksrc, st, mode);
  };
  packL(Wih_pk, gru_Wih, NT_G, KC_IH, 768, 129, 129, 0);
  packL(Whh_pk, gru_Whh, NT_G, KC_HH, 768, 256, 256, 0);
  packL(pW1_pk, pW1, NT_H, KC_W1, 256, 641, 641, 2);
  packL(pW2_pk, pW2, NT_H, KC_W2, 256, 256, 256, 0);
  packL(pW3_pk, pW3, 1,    KC_W3, NOUTV, 256, 256, 0);
  packL(fW1_pk, fW1, NT_H, KC_W1, 256, 641, 641, 2);
  packL(fW2_pk, fW2, NT_H, KC_W2, 256, 256, 256, 0);
  packL(fW3_pk, fW3, 1,    KC_W3, NOUTV, 256, 256, 0);
  prep_y<<<(N_SEQ * NIT + 255) / 256, 256, 0, stream>>>(Y, epw, dttw);

  // dynamic LDS: bf16 blocks + float scratch
  size_t smem = (size_t)(2 * MTILE * S_H + MTILE * S_XG + MTILE * S_XM + 2 * MTILE * S_A) * 2
              + (size_t)(MTILE * DD + 2 * DD + 2 * MTILE + MTILE * 16) * 4;   // 116480 B
  static int attr_set = 0;  // host-side config, not a stream op; deterministic
  if (!attr_set) {
    hipFuncSetAttribute((const void*)semiar_main,
                        hipFuncAttributeMaxDynamicSharedMemorySize, (int)smem);
    attr_set = 1;
  }

  semiar_main<<<2 * NBLK, 256, smem, stream>>>(
      TE_w, shared_w, shared_b, TE_lam,
      gru_bih, gru_bhh, pb1, pb2, pb3, fb1, fb2, fb3,
      h_int, dt_init,
      Wih_pk, Whh_pk, pW1_pk, pW2_pk, pW3_pk, fW1_pk, fW2_pk, fW3_pk,
      epw, dttw, out);
}
